// DynamicSpatialAttention_47485158425355
// MI455X (gfx1250) — compile-verified, matrix operand order checked by reading, not run on hardware
//
#include <hip/hip_runtime.h>
#include <cstdint>

// ---------------------------------------------------------------------------
// DynamicSpatialAttention for MI455X (gfx1250).
// Bandwidth-bound: ~268MB mandatory HBM traffic vs ~0.5 GFLOP of conv math.
// Strategy: two passes over x (L2-resident between passes, 134MB < 192MB L2),
// NT stores for Y, async global->LDS staging for conv tiles, WMMA for the
// one real matmul (router layer 1).
// ---------------------------------------------------------------------------

typedef __attribute__((ext_vector_type(4)))  float    f4;
typedef __attribute__((ext_vector_type(16))) _Float16 v16h;
typedef __attribute__((ext_vector_type(8)))  float    v8f;

#define B_   32
#define C_   64
#define H_   128
#define W_   128
#define HW   16384          // 128*128
#define PW   144            // padded map row stride (floats), 16B-friendly
#define PROWS 140           // 6 + 128 + 6 rows
#define MAPSZ (PW*PROWS)    // 20160 floats per map per batch

// workspace layout (float indices)
static constexpr size_t WS_MAPS = 0;                          // 2*32*20160
static constexpr size_t WS_PART = (size_t)2*B_*MAPSZ;         // 32*16*2 partial sums
static constexpr size_t WS_RW   = WS_PART + 1024;             // 32*4 router weights
static constexpr size_t WS_T    = WS_RW + 128;                // temperature
static constexpr size_t WS_WEFF = WS_T + 8;                   // 32*256 folded taps
static constexpr size_t WS_CB   = WS_WEFF + (size_t)B_*256;   // 3*HW const maps
static constexpr size_t SA_OFF  = (size_t)B_*C_*HW;           // sa offset in d_out

// --------------------------- K0: zero padded maps --------------------------
__global__ void k0_zero(float* __restrict__ ws) {
  size_t i = ((size_t)blockIdx.x * blockDim.x + threadIdx.x) * 4;
  if (i < WS_PART) { f4 z = {0.f,0.f,0.f,0.f}; *(f4*)(ws + i) = z; }
}

// ------------- K1: channel avg/max -> padded maps + partial sums ------------
__global__ void __launch_bounds__(256) k1_reduce(const float* __restrict__ x,
                                                 float* __restrict__ ws) {
  int blk = blockIdx.x;          // b*16 + chunk
  int b = blk >> 4, chunk = blk & 15;
  int t = threadIdx.x;
  int p0 = chunk * 1024 + t * 4; // 4 consecutive pixels, same row (W%4==0)
  const float* xb = x + (size_t)b * C_ * HW;
  f4 s = {0.f,0.f,0.f,0.f};
  f4 m = {-3.402823466e38f,-3.402823466e38f,-3.402823466e38f,-3.402823466e38f};
  for (int c = 0; c < C_; ++c) {
    f4 v = *(const f4*)(xb + (size_t)c * HW + p0);   // coalesced, RT -> L2 resident
    s += v;
    m.x = fmaxf(m.x, v.x); m.y = fmaxf(m.y, v.y);
    m.z = fmaxf(m.z, v.z); m.w = fmaxf(m.w, v.w);
  }
  f4 a = s * (1.f / (float)C_);
  int h = p0 >> 7, w = p0 & 127;
  size_t off = (size_t)(h + 6) * PW + 8 + w;        // padded, 16B aligned
  *(f4*)(ws + WS_MAPS + (size_t)b * MAPSZ + off)        = a;
  *(f4*)(ws + WS_MAPS + (size_t)(B_ + b) * MAPSZ + off) = m;

  __shared__ float rA[256], rM[256];
  rA[t] = a.x + a.y + a.z + a.w;
  rM[t] = m.x + m.y + m.z + m.w;
  __syncthreads();
  for (int o = 128; o > 0; o >>= 1) {               // deterministic tree reduce
    if (t < o) { rA[t] += rA[t + o]; rM[t] += rM[t + o]; }
    __syncthreads();
  }
  if (t == 0) {
    ws[WS_PART + (size_t)blk * 2 + 0] = rA[0];
    ws[WS_PART + (size_t)blk * 2 + 1] = rM[0];
  }
}

// ---- K2: router MLP (layer1 via v_wmma_f32_16x16x32_f16) + weight folding --
__global__ void k2_router(float* __restrict__ ws,
    const float* __restrict__ dw0, const float* __restrict__ dw1,
    const float* __restrict__ dw2,
    const float* __restrict__ pw0w, const float* __restrict__ pw1w,
    const float* __restrict__ pw2w,
    const float* __restrict__ rw1, const float* __restrict__ rb1,
    const float* __restrict__ rw2, const float* __restrict__ rb2,
    const float* __restrict__ traw) {
  int lane = threadIdx.x;                 // exactly one wave32
  __shared__ float pool[32][2];
  __shared__ float hid[32][8];
  __shared__ float rwS[32][3];

  float sA = 0.f, sM = 0.f;
  for (int k = 0; k < 16; ++k) {          // fixed-order: deterministic
    sA += ws[WS_PART + (size_t)(lane * 16 + k) * 2 + 0];
    sM += ws[WS_PART + (size_t)(lane * 16 + k) * 2 + 1];
  }
  pool[lane][0] = sA * (1.f / (float)HW);
  pool[lane][1] = sM * (1.f / (float)HW); // pooled xg/yg are exactly 0
  __syncthreads();

  // B (32x16 f16): lane n<16 holds K=0..7,16..23 for column n; lanes>=16 hold K=8..15,24..31 (pad=0)
  v16h Bm = {};
  if (lane < 16 && lane < 8) {
    for (int k = 0; k < 4; ++k) Bm[k] = (_Float16)rw1[lane * 4 + k];
  }
  // hid[m][n] = sum_k pooled[m][k] * rw1[n][k] ; two WMMAs cover 32 batches
  for (int half = 0; half < 2; ++half) {
    int base = half * 16;
    // A (16x32 f16): lanes 0-15 hold K=0..7,16..23 for row M=lane; only K<2 nonzero
    v16h Am = {};
    if (lane < 16) {
      Am[0] = (_Float16)pool[base + lane][0];
      Am[1] = (_Float16)pool[base + lane][1];
    }
    v8f Cz = {};
    v8f D = __builtin_amdgcn_wmma_f32_16x16x32_f16(false, Am, false, Bm,
                                                   (short)0, Cz, false, false);
    int nn = lane & 15;
    int hiAdd = (lane >> 4) * 8;          // lanes 16-31 carry rows M=j+8
    if (nn < 8) {
      #pragma unroll
      for (int j = 0; j < 8; ++j)
        hid[base + j + hiAdd][nn] = fmaxf(D[j] + rb1[nn], 0.f);
    }
  }
  __syncthreads();

  // layer 2 + softmax, one batch per lane
  float lg[3];
  #pragma unroll
  for (int o = 0; o < 3; ++o) {
    float a = rb2[o];
    #pragma unroll
    for (int n = 0; n < 8; ++n) a += rw2[o * 8 + n] * hid[lane][n];
    lg[o] = a;
  }
  float mx = fmaxf(lg[0], fmaxf(lg[1], lg[2]));
  float e0 = expf(lg[0] - mx), e1 = expf(lg[1] - mx), e2 = expf(lg[2] - mx);
  float inv = 1.f / (e0 + e1 + e2);
  rwS[lane][0] = e0 * inv; rwS[lane][1] = e1 * inv; rwS[lane][2] = e2 * inv;
  ws[WS_RW + (size_t)lane * 4 + 0] = e0 * inv;
  ws[WS_RW + (size_t)lane * 4 + 1] = e1 * inv;
  ws[WS_RW + (size_t)lane * 4 + 2] = e2 * inv;
  if (lane == 0) ws[WS_T] = log1pf(expf(traw[0])) + 1e-6f;
  __syncthreads();

  // fold dw (3 dilated kernels) * pw * router weight into one 11x11 union-tap
  // kernel per batch per channel (avg, max)
  const int OFFa[11] = {-6,-4,-3,-2,-1,0,1,2,3,4,6};
  for (int idx = lane; idx < 32 * 242; idx += 32) {
    int b = idx / 242, r2 = idx % 242;
    int c = r2 / 121, tt = r2 % 121;
    int dy = OFFa[tt / 11], dx = OFFa[tt % 11];
    float v = 0.f;
    if (dy >= -1 && dy <= 1 && dx >= -1 && dx <= 1)
      v += rwS[b][0] * pw0w[c] * dw0[c * 9 + (dy + 1) * 3 + (dx + 1)];
    if (dy >= -3 && dy <= 3 && dx >= -3 && dx <= 3)
      v += rwS[b][1] * pw1w[c] * dw1[c * 49 + (dy + 3) * 7 + (dx + 3)];
    if (((dy & 1) == 0) && ((dx & 1) == 0))
      v += rwS[b][2] * pw2w[c] * dw2[c * 49 + (dy / 2 + 3) * 7 + (dx / 2 + 3)];
    ws[WS_WEFF + (size_t)b * 256 + c * 128 + tt] = v;
  }
}

// ----- K2b: batch-independent const maps from analytic xg/yg channels -------
__global__ void k2b_constmaps(
    const float* __restrict__ dw0, const float* __restrict__ dw1,
    const float* __restrict__ dw2,
    const float* __restrict__ pw0w, const float* __restrict__ pw0b,
    const float* __restrict__ pw1w, const float* __restrict__ pw1b,
    const float* __restrict__ pw2w, const float* __restrict__ pw2b,
    float* __restrict__ ws) {
  int j = blockIdx.x * 256 + threadIdx.x;   // 0 .. 3*HW-1
  int br = j >> 14;
  int p  = j & (HW - 1);
  int h = p >> 7, w = p & 127;
  const float* dw; const float* pw; const float* pb; int k, d, pad;
  if (br == 0)      { dw = dw0; pw = pw0w; pb = pw0b; k = 3; d = 1; pad = 1; }
  else if (br == 1) { dw = dw1; pw = pw1w; pb = pw1b; k = 7; d = 1; pad = 3; }
  else              { dw = dw2; pw = pw2w; pb = pw2b; k = 7; d = 2; pad = 6; }
  int kk = k * k;
  float cx = 0.f, cy = 0.f;                 // conv(xg), conv(yg) with zero pad
  for (int i = 0; i < k; ++i) {
    int yy = h + d * i - pad;
    if (yy < 0 || yy >= H_) continue;
    float ygv = -1.f + 2.f * (float)yy * (1.f / 127.f);
    for (int jx = 0; jx < k; ++jx) {
      int xx = w + d * jx - pad;
      if (xx < 0 || xx >= W_) continue;
      float xgv = -1.f + 2.f * (float)xx * (1.f / 127.f);
      cx += dw[2 * kk + i * k + jx] * xgv;
      cy += dw[3 * kk + i * k + jx] * ygv;
    }
  }
  ws[WS_CB + (size_t)br * HW + p] = pw[2] * cx + pw[3] * cy + pb[0];
}

// -------- K3: async-LDS staging, fused conv + sigmoid + x*sa (NT stores) ----
__global__ void __launch_bounds__(256) k3_main(const float* __restrict__ x,
                                               const float* __restrict__ ws,
                                               float* __restrict__ out) {
  __shared__ float sT[2 * 20 * PW];   // 2 maps x (8+12) halo rows x 144 cols
  __shared__ float saT[1024];
  __shared__ float wE[256];
  __shared__ float rwT[4];

  int blk = blockIdx.x;               // b*16 + ty ; tile = 8 rows x 128 cols
  int b = blk >> 4, ty = blk & 15;
  int t = threadIdx.x;

  // Stage padded avg/max tiles via CDNA5 async global->LDS (ASYNCcnt path).
  const float* maps = ws + WS_MAPS;
  for (int k = t; k < 1440; k += 256) {        // 2 maps * 20 rows * 36 float4
    int m = k / 720, rem = k % 720;
    int row = rem / 36, q = rem % 36;
    const float* gp = maps + (size_t)(m * B_ + b) * MAPSZ +
                      (size_t)(ty * 8 + row) * PW + q * 4;
    unsigned long long ga = (unsigned long long)(uintptr_t)gp;
    unsigned int la = (unsigned int)(uintptr_t)(&sT[m * 2880 + row * PW + q * 4]);
    asm volatile("global_load_async_to_lds_b128 %0, %1, off"
                 :: "v"(la), "v"(ga) : "memory");
  }
  {
    const float* gp = ws + WS_WEFF + (size_t)b * 256 + t;
    unsigned long long ga = (unsigned long long)(uintptr_t)gp;
    unsigned int la = (unsigned int)(uintptr_t)(&wE[t]);
    asm volatile("global_load_async_to_lds_b32 %0, %1, off"
                 :: "v"(la), "v"(ga) : "memory");
  }
  if (t < 3)  rwT[t] = ws[WS_RW + (size_t)b * 4 + t];
  if (t == 3) rwT[3] = ws[WS_T];
  asm volatile("s_wait_asynccnt 0x0" ::: "memory");
  __syncthreads();

  float rw0 = rwT[0], rw1v = rwT[1], rw2v = rwT[2];
  float invT = 1.f / rwT[3];

  float acc[4]; int pb_[4];
  #pragma unroll
  for (int i = 0; i < 4; ++i) {
    int p = i * 256 + t;
    int r = p >> 7, w = p & 127;
    int gpx = ty * 1024 + p;
    acc[i] = rw0 * ws[WS_CB + gpx] + rw1v * ws[WS_CB + HW + gpx] +
             rw2v * ws[WS_CB + 2 * HW + gpx];
    pb_[i] = (r + 6) * PW + 8 + w;
  }
  const int OFFa[11] = {-6,-4,-3,-2,-1,0,1,2,3,4,6};
  #pragma unroll
  for (int c = 0; c < 2; ++c) {
    #pragma unroll
    for (int iy = 0; iy < 11; ++iy) {
      #pragma unroll
      for (int ix = 0; ix < 11; ++ix) {
        int dy = OFFa[iy], dx = OFFa[ix];
        // taps covered by any branch (dense 7x7 union even 7x7); rest fold to 0
        bool cov = (dy >= -3 && dy <= 3 && dx >= -3 && dx <= 3) ||
                   (((dy & 1) == 0) && ((dx & 1) == 0));
        if (cov) {
          float wv = wE[c * 128 + iy * 11 + ix];
          int dd = c * 2880 + dy * PW + dx;
          #pragma unroll
          for (int i = 0; i < 4; ++i) acc[i] += wv * sT[pb_[i] + dd];
        }
      }
    }
  }
  #pragma unroll
  for (int i = 0; i < 4; ++i) {
    float sa = 1.f / (1.f + expf(-acc[i] * invT));
    saT[i * 256 + t] = sa;
    out[SA_OFF + (size_t)b * HW + ty * 1024 + i * 256 + t] = sa;
  }
  __syncthreads();

  // Y = x * sa : x re-read (L2-resident from K1), Y streamed out non-temporal
  f4 s4 = *(const f4*)&saT[t * 4];
  #pragma unroll 4
  for (int c = 0; c < C_; ++c) {
    size_t e = ((size_t)(b * C_ + c) << 14) + (size_t)ty * 1024 + (size_t)t * 4;
    f4 v = *(const f4*)(x + e);
    f4 r; r.x = v.x * s4.x; r.y = v.y * s4.y; r.z = v.z * s4.z; r.w = v.w * s4.w;
    __builtin_nontemporal_store(r, (f4*)(out + e));
  }
}

// ---------------------------------------------------------------------------
extern "C" void kernel_launch(void* const* d_in, const int* in_sizes, int n_in,
                              void* d_out, int out_size, void* d_ws, size_t ws_size,
                              hipStream_t stream) {
  (void)in_sizes; (void)n_in; (void)out_size; (void)ws_size;
  const float* x     = (const float*)d_in[0];
  const float* dw0   = (const float*)d_in[1];
  const float* dw1   = (const float*)d_in[2];
  const float* dw2   = (const float*)d_in[3];
  const float* pw0w  = (const float*)d_in[4];
  const float* pw0b  = (const float*)d_in[5];
  const float* pw1w  = (const float*)d_in[6];
  const float* pw1b  = (const float*)d_in[7];
  const float* pw2w  = (const float*)d_in[8];
  const float* pw2b  = (const float*)d_in[9];
  const float* rw1   = (const float*)d_in[10];
  const float* rb1   = (const float*)d_in[11];
  const float* rw2   = (const float*)d_in[12];
  const float* rb2   = (const float*)d_in[13];
  const float* traw  = (const float*)d_in[14];
  float* out = (float*)d_out;
  float* ws  = (float*)d_ws;

  k0_zero<<<1260, 256, 0, stream>>>(ws);                       // zero padded maps
  k1_reduce<<<B_ * 16, 256, 0, stream>>>(x, ws);               // pass 1 over x
  k2_router<<<1, 32, 0, stream>>>(ws, dw0, dw1, dw2, pw0w, pw1w, pw2w,
                                  rw1, rb1, rw2, rb2, traw);   // WMMA router
  k2b_constmaps<<<192, 256, 0, stream>>>(dw0, dw1, dw2, pw0w, pw0b,
                                         pw1w, pw1b, pw2w, pw2b, ws);
  k3_main<<<B_ * 16, 256, 0, stream>>>(x, ws, out);            // pass 2 + output
}
